// TransformerEncoderLayer_26319559590770
// MI455X (gfx1250) — matrix operand order repaired and checked by reading, not compiled, not run on hardware
//
#include <hip/hip_runtime.h>
#include <hip/hip_bf16.h>
#include <math.h>
#include <stdint.h>

// ---------------------------------------------------------------------------
// Transformer encoder layer (B=8, L=2048, D=512, h=8, d=64, DF=2048) for
// gfx1250. All GEMM-shaped work runs on v_wmma_f32_16x16x32_f16 (f32 accum).
// GEMM staging uses CDNA5 async global->LDS copies with double buffering.
// ---------------------------------------------------------------------------

typedef __attribute__((ext_vector_type(16))) _Float16 v16h;
typedef __attribute__((ext_vector_type(8)))  _Float16 v8h;
typedef __attribute__((ext_vector_type(4)))  _Float16 v4h;
typedef __attribute__((ext_vector_type(8)))  float    v8f;
typedef __attribute__((ext_vector_type(4)))  float    v4f;

#define LN_EPS 1e-9f

static __device__ __forceinline__ v16h hcat(v8h lo, v8h hi) {
  return __builtin_shufflevector(lo, hi, 0,1,2,3,4,5,6,7,8,9,10,11,12,13,14,15);
}
static __device__ __forceinline__ v8f wmma16(v16h a, v16h b, v8f c) {
  // D = A(16x32 f16) * B(32x16 f16) + C(16x16 f32)
  return __builtin_amdgcn_wmma_f32_16x16x32_f16(false, a, false, b, (short)0, c,
                                                false, false);
}
// Async global->LDS copy (ASYNCcnt-tracked). ldsaddr = low 32 bits of the
// generic shared pointer (aperture mapping: addr[31:0] is the LDS offset).
static __device__ __forceinline__ void async_ld128(void* lds, const void* gptr) {
  unsigned la = (unsigned)(uintptr_t)lds;
  asm volatile("global_load_async_to_lds_b128 %0, %1, off"
               :: "v"(la), "v"(gptr) : "memory");
}

// ---------------------------------------------------------------------------
// Prep kernels
// ---------------------------------------------------------------------------
__global__ void k_cvt4(const float* __restrict__ s, _Float16* __restrict__ d, int n) {
  int i = (blockIdx.x * 256 + threadIdx.x) * 4;
  if (i < n) {
    v4f v = *(const v4f*)(s + i);
    v4h h;
    #pragma unroll
    for (int j = 0; j < 4; ++j) h[j] = (_Float16)v[j];
    *(v4h*)(d + i) = h;
  }
}

// Wt[n*K + k] = (f16) W[k*N + n]   (W is KxN row-major -> Wt is NxK row-major)
__global__ void k_transpose_cvt(const float* __restrict__ W, _Float16* __restrict__ Wt,
                                int K, int N) {
  int i = blockIdx.x * 256 + threadIdx.x;
  if (i < K * N) {
    int n = i / K, k = i - n * K;
    Wt[i] = (_Float16)W[(size_t)k * N + n];
  }
}

__global__ void k_concat_bias(const float* __restrict__ bq, const float* __restrict__ bk,
                              const float* __restrict__ bv, float* __restrict__ o) {
  int i = blockIdx.x * 256 + threadIdx.x;
  if (i < 512)       o[i] = bq[i];
  else if (i < 1024) o[i] = bk[i - 512];
  else if (i < 1536) o[i] = bv[i - 1024];
}

// ---------------------------------------------------------------------------
// WMMA GEMM:  C[M,N] = A[M,K](f16) * Bt[N,K](f16)^T + bias
//   block tile 128x128, 8 waves (2x4), wave tile 64x32, K-step 32,
//   double-buffered async global->LDS staging.
// MODE 1: Cf16 = relu(C)            (FFN first layer)
// MODE 2: Cf32 = C + Res (fp32)     (pre-layernorm residual add)
// MODE 3: scatter fused QKV -> Qh[b,h,l,d], Kh[b,h,l,d], Vt[b,h,d,l] (f16)
// ---------------------------------------------------------------------------
#define BK 32
#define LPAD 8

template <int MODE>
__global__ __launch_bounds__(256) void k_gemm_wmma(
    const _Float16* __restrict__ A, const _Float16* __restrict__ Bt,
    const float* __restrict__ bias, const float* __restrict__ Res,
    float* __restrict__ Cf32, _Float16* __restrict__ Cf16,
    _Float16* __restrict__ Qh, _Float16* __restrict__ Kh, _Float16* __restrict__ Vt,
    int M, int N, int K) {
  __shared__ _Float16 sA[2][128][BK + LPAD];
  __shared__ _Float16 sB[2][128][BK + LPAD];

  const int tid  = threadIdx.x;
  const int lane = tid & 31, wave = tid >> 5;
  const int hi = lane >> 4, ln = lane & 15;
  const int wm = (wave & 1) * 64;   // 2 wave-rows of 64
  const int wn = (wave >> 1) * 32;  // 4 wave-cols of 32
  const int tileM = blockIdx.y * 128;
  const int tileN = blockIdx.x * 128;

  // This thread's two staging chunks (128x32 halfs = 512 16B chunks, 2/thread)
  const int r0 = tid >> 2,           off0 = (tid & 3) * 8;
  const int r1 = (tid + 256) >> 2,   off1 = ((tid + 256) & 3) * 8;

  // 4 async b128 issues per stage per thread (A0,B0,A1,B1)
  auto stage = [&](int buf, int k0) {
    async_ld128(&sA[buf][r0][off0], A  + (size_t)(tileM + r0) * K + k0 + off0);
    async_ld128(&sB[buf][r0][off0], Bt + (size_t)(tileN + r0) * K + k0 + off0);
    async_ld128(&sA[buf][r1][off1], A  + (size_t)(tileM + r1) * K + k0 + off1);
    async_ld128(&sB[buf][r1][off1], Bt + (size_t)(tileN + r1) * K + k0 + off1);
  };

  v8f acc[4][2] = {};
  stage(0, 0);
  int buf = 0;

  for (int k0 = 0; k0 < K; k0 += BK) {
    const bool more = (k0 + BK) < K;
    if (more) {
      stage(buf ^ 1, k0 + BK);
      // 8 outstanding; async loads complete in order -> <=4 means prev batch done
      asm volatile("s_wait_asynccnt 4" ::: "memory");
    } else {
      asm volatile("s_wait_asynccnt 0" ::: "memory");
    }
    __syncthreads();

    // ---- fragments + 8 WMMAs ----------------------------------------------
    v16h fa[4], fb[2];
    #pragma unroll
    for (int i = 0; i < 4; ++i) {
      const _Float16* ap = &sA[buf][wm + i * 16 + ln][0];
      fa[i] = hcat(*(const v8h*)(ap + hi * 16), *(const v8h*)(ap + hi * 16 + 8));
    }
    #pragma unroll
    for (int j = 0; j < 2; ++j) {
      const _Float16* bp = &sB[buf][wn + j * 16 + ln][0];
      fb[j] = hcat(*(const v8h*)(bp + hi * 16), *(const v8h*)(bp + hi * 16 + 8));
    }
    #pragma unroll
    for (int i = 0; i < 4; ++i)
      #pragma unroll
      for (int j = 0; j < 2; ++j)
        acc[i][j] = wmma16(fa[i], fb[j], acc[i][j]);
    __syncthreads();   // all waves done reading buf before it is re-staged
    buf ^= 1;
  }

  // ---- epilogue (C layout: reg r -> row r+8*hi, col = ln) -------------------
  #pragma unroll
  for (int i = 0; i < 4; ++i) {
    #pragma unroll
    for (int j = 0; j < 2; ++j) {
      const int gm0 = tileM + wm + i * 16 + hi * 8;
      const int gn  = tileN + wn + j * 16 + ln;
      const float bb = bias[gn];
      if (MODE == 1) {
        #pragma unroll
        for (int r = 0; r < 8; ++r) {
          float v = acc[i][j][r] + bb;
          v = v > 0.f ? v : 0.f;
          Cf16[(size_t)(gm0 + r) * N + gn] = (_Float16)v;
        }
      } else if (MODE == 2) {
        #pragma unroll
        for (int r = 0; r < 8; ++r) {
          size_t idx = (size_t)(gm0 + r) * N + gn;
          Cf32[idx] = acc[i][j][r] + bb + Res[idx];
        }
      } else {  // MODE 3: fused QKV scatter
        const int which = gn >> 9;           // 0=Q 1=K 2=V
        const int cc = gn & 511;
        const int hh = cc >> 6, dd = cc & 63;
        #pragma unroll
        for (int r = 0; r < 8; ++r) {
          int m = gm0 + r;                   // = b*L + l
          int bI = m >> 11, l = m & 2047;
          size_t bh = (size_t)(bI * 8 + hh);
          _Float16 hv = (_Float16)(acc[i][j][r] + bb);
          if (which == 0)      Qh[(bh * 2048 + l) * 64 + dd] = hv;
          else if (which == 1) Kh[(bh * 2048 + l) * 64 + dd] = hv;
          else                 Vt[(bh * 64 + dd) * 2048 + l] = hv;
        }
      }
    }
  }
}

// ---------------------------------------------------------------------------
// Flash attention: one wave per 16-row Q tile; streams 32 keys/iter.
//   S = Q Kt (4 wmma), online softmax, P through LDS -> A-layout, ctx += P V
// Ctx written back as f16 [B, L, D] (D = h*64) for the Wo GEMM.
// ---------------------------------------------------------------------------
__global__ __launch_bounds__(256) void k_attn_flash(
    const _Float16* __restrict__ Qh, const _Float16* __restrict__ Kh,
    const _Float16* __restrict__ Vt, const unsigned char* __restrict__ mask,
    _Float16* __restrict__ Ctx) {
  __shared__ _Float16 sP[8][16][BK + LPAD];  // per-wave P staging (10 KB)

  const int tid = threadIdx.x, lane = tid & 31, wave = tid >> 5;
  const int hi = lane >> 4, ln = lane & 15;
  const int gw = blockIdx.x * 8 + wave;      // 8192 wave-tiles total
  const int qt = gw & 127;                   // q tile (L/16 = 128)
  const int bh = gw >> 7;                    // 0..63
  const int b = bh >> 3, h = bh & 7;

  const _Float16* Qp = Qh + ((size_t)bh * 2048 + qt * 16) * 64;
  const _Float16* Kp = Kh + (size_t)bh * 2048 * 64;
  const _Float16* Vp = Vt + (size_t)bh * 64 * 2048;
  const unsigned char* mp = mask + (size_t)b * 2048;

  // Q A-fragments (K = d: two steps of 32)
  v16h qa[2];
  #pragma unroll
  for (int s = 0; s < 2; ++s) {
    const _Float16* qr = Qp + ln * 64 + s * 32;
    qa[s] = hcat(*(const v8h*)(qr + hi * 16), *(const v8h*)(qr + hi * 16 + 8));
  }

  float mrun[8], lrun[8];
  v8f o[4] = {};
  #pragma unroll
  for (int r = 0; r < 8; ++r) { mrun[r] = -__builtin_inff(); lrun[r] = 0.f; }

  for (int kb = 0; kb < 2048; kb += 32) {
    // ---- scores for 32 keys: two 16-key C tiles -----------------------------
    v8f s2[2];
    #pragma unroll
    for (int sub = 0; sub < 2; ++sub) {
      const int key = kb + sub * 16 + ln;
      const _Float16* kr = Kp + (size_t)key * 64;
      v16h kb0 = hcat(*(const v8h*)(kr + hi * 16),      *(const v8h*)(kr + hi * 16 + 8));
      v16h kb1 = hcat(*(const v8h*)(kr + 32 + hi * 16), *(const v8h*)(kr + 32 + hi * 16 + 8));
      v8f z = {};
      z = wmma16(qa[1], kb1, z);
      z = wmma16(qa[0], kb0, z);
      const bool msk = mp[key] != 0;
      #pragma unroll
      for (int r = 0; r < 8; ++r) z[r] = msk ? -1e9f : z[r] * 0.125f;  // d^-0.5
      s2[sub] = z;
    }

    // ---- online softmax (row r+8*hi lives in reg r across a 16-lane group) --
    #pragma unroll
    for (int r = 0; r < 8; ++r) {
      float bm = fmaxf(s2[0][r], s2[1][r]);
      #pragma unroll
      for (int x = 1; x < 16; x <<= 1) bm = fmaxf(bm, __shfl_xor(bm, x, 32));
      const float mn   = fmaxf(mrun[r], bm);
      const float corr = __expf(mrun[r] - mn);
      const float p0 = __expf(s2[0][r] - mn);
      const float p1 = __expf(s2[1][r] - mn);
      float ps = p0 + p1;
      #pragma unroll
      for (int x = 1; x < 16; x <<= 1) ps += __shfl_xor(ps, x, 32);
      lrun[r] = lrun[r] * corr + ps;
      mrun[r] = mn;
      #pragma unroll
      for (int g = 0; g < 4; ++g) o[g][r] *= corr;
      const int row = r + hi * 8;
      sP[wave][row][ln]      = (_Float16)p0;
      sP[wave][row][16 + ln] = (_Float16)p1;
    }

    // wave-local LDS bounce: C-layout P -> A-layout fragment
    asm volatile("s_wait_dscnt 0" ::: "memory");
    const _Float16* pp = &sP[wave][ln][0];
    v16h pa = hcat(*(const v8h*)(pp + hi * 16), *(const v8h*)(pp + hi * 16 + 8));

    // ---- ctx += P * V (V pre-transposed: rows are d, contiguous over keys) --
    #pragma unroll
    for (int g = 0; g < 4; ++g) {
      const _Float16* vr = Vp + (size_t)(g * 16 + ln) * 2048 + kb + hi * 16;
      v16h vb = hcat(*(const v8h*)vr, *(const v8h*)(vr + 8));
      o[g] = wmma16(pa, vb, o[g]);
    }
    asm volatile("s_wait_dscnt 0" ::: "memory");  // before next iter rewrites sP
  }

  // ---- normalize and write ctx f16 [B, L, h*64] -----------------------------
  #pragma unroll
  for (int r = 0; r < 8; ++r) {
    const float inv = 1.f / lrun[r];
    const int l = qt * 16 + r + hi * 8;
    const size_t base = ((size_t)b * 2048 + l) * 512 + h * 64;
    #pragma unroll
    for (int g = 0; g < 4; ++g)
      Ctx[base + g * 16 + ln] = (_Float16)(o[g][r] * inv);
  }
}

// ---------------------------------------------------------------------------
// LayerNorm over rows of 512: 256 threads/row, 2 elems/thread.
// ---------------------------------------------------------------------------
__global__ __launch_bounds__(256) void k_layernorm(
    const float* __restrict__ X, const float* __restrict__ g,
    const float* __restrict__ be, float* __restrict__ Y,
    _Float16* __restrict__ Yh) {
  __shared__ float rsum[8], rsq[8];
  const int row = blockIdx.x, tid = threadIdx.x;
  const float* x = X + (size_t)row * 512;
  const float v0 = x[tid], v1 = x[tid + 256];
  float s = v0 + v1, q = v0 * v0 + v1 * v1;
  #pragma unroll
  for (int m = 1; m < 32; m <<= 1) { s += __shfl_xor(s, m, 32); q += __shfl_xor(q, m, 32); }
  if ((tid & 31) == 0) { rsum[tid >> 5] = s; rsq[tid >> 5] = q; }
  __syncthreads();
  float ts = 0.f, tq = 0.f;
  #pragma unroll
  for (int i = 0; i < 8; ++i) { ts += rsum[i]; tq += rsq[i]; }
  const float mean = ts * (1.f / 512.f);
  const float var  = fmaxf(tq * (1.f / 512.f) - mean * mean, 0.f);
  const float rs   = rsqrtf(var + LN_EPS);
  const float y0 = (v0 - mean) * rs * g[tid] + be[tid];
  const float y1 = (v1 - mean) * rs * g[tid + 256] + be[tid + 256];
  Y[(size_t)row * 512 + tid] = y0;
  Y[(size_t)row * 512 + tid + 256] = y1;
  if (Yh) {
    Yh[(size_t)row * 512 + tid] = (_Float16)y0;
    Yh[(size_t)row * 512 + tid + 256] = (_Float16)y1;
  }
}

// ---------------------------------------------------------------------------
// Host orchestration
// ---------------------------------------------------------------------------
extern "C" void kernel_launch(void* const* d_in, const int* in_sizes, int n_in,
                              void* d_out, int out_size, void* d_ws, size_t ws_size,
                              hipStream_t stream) {
  (void)in_sizes; (void)n_in; (void)out_size; (void)ws_size;
  const float* feats = (const float*)d_in[0];
  const unsigned char* mask = (const unsigned char*)d_in[1];  // bool array
  const float* Wq = (const float*)d_in[2];  const float* bq = (const float*)d_in[3];
  const float* Wk = (const float*)d_in[4];  const float* bk = (const float*)d_in[5];
  const float* Wv = (const float*)d_in[6];  const float* bv = (const float*)d_in[7];
  const float* Wo = (const float*)d_in[8];  const float* bo = (const float*)d_in[9];
  const float* W1 = (const float*)d_in[10]; const float* b1 = (const float*)d_in[11];
  const float* W2 = (const float*)d_in[12]; const float* b2 = (const float*)d_in[13];
  const float* g1 = (const float*)d_in[14]; const float* be1 = (const float*)d_in[15];
  const float* g2 = (const float*)d_in[16]; const float* be2 = (const float*)d_in[17];

  const size_t ML = 8 * 2048;  // 16384 rows
  size_t off = 0;
  auto carve = [&](size_t bytes) -> void* {
    void* p = (char*)d_ws + off;
    off += (bytes + 255) & ~(size_t)255;
    return p;
  };
  _Float16* Xh     = (_Float16*)carve(ML * 512 * 2);
  _Float16* Wqkv_t = (_Float16*)carve(1536 * 512 * 2);
  _Float16* Wo_t   = (_Float16*)carve(512 * 512 * 2);
  _Float16* W1_t   = (_Float16*)carve((size_t)2048 * 512 * 2);
  _Float16* W2_t   = (_Float16*)carve((size_t)512 * 2048 * 2);
  float*    bqkv   = (float*)carve(1536 * 4);
  _Float16* Qh     = (_Float16*)carve(ML * 512 * 2);
  _Float16* Kh     = (_Float16*)carve(ML * 512 * 2);
  _Float16* Vt     = (_Float16*)carve(ML * 512 * 2);
  _Float16* Ctxh   = (_Float16*)carve(ML * 512 * 2);
  float*    pre    = (float*)carve(ML * 512 * 4);     // reused (pre-LN1, pre-LN2)
  float*    x32    = (float*)carve(ML * 512 * 4);
  _Float16* xh     = (_Float16*)carve(ML * 512 * 2);
  _Float16* Hh     = (_Float16*)carve(ML * 2048 * 2);

  // prep
  k_cvt4<<<(int)(ML * 512 / 4 / 256), 256, 0, stream>>>(feats, Xh, (int)(ML * 512));
  k_transpose_cvt<<<1024, 256, 0, stream>>>(Wq, Wqkv_t,               512, 512);
  k_transpose_cvt<<<1024, 256, 0, stream>>>(Wk, Wqkv_t + 512 * 512,   512, 512);
  k_transpose_cvt<<<1024, 256, 0, stream>>>(Wv, Wqkv_t + 1024 * 512,  512, 512);
  k_transpose_cvt<<<1024, 256, 0, stream>>>(Wo, Wo_t,                 512, 512);
  k_transpose_cvt<<<4096, 256, 0, stream>>>(W1, W1_t,                 512, 2048);
  k_transpose_cvt<<<4096, 256, 0, stream>>>(W2, W2_t,                 2048, 512);
  k_concat_bias<<<6, 256, 0, stream>>>(bq, bk, bv, bqkv);

  // fused QKV projection -> Q,K row-major, V transposed
  k_gemm_wmma<3><<<dim3(12, 128), 256, 0, stream>>>(
      Xh, Wqkv_t, bqkv, nullptr, nullptr, nullptr, Qh, Kh, Vt, 16384, 1536, 512);

  // attention
  k_attn_flash<<<1024, 256, 0, stream>>>(Qh, Kh, Vt, mask, Ctxh);

  // output projection + residual, then LN1
  k_gemm_wmma<2><<<dim3(4, 128), 256, 0, stream>>>(
      Ctxh, Wo_t, bo, feats, pre, nullptr, nullptr, nullptr, nullptr, 16384, 512, 512);
  k_layernorm<<<16384, 256, 0, stream>>>(pre, g1, be1, x32, xh);

  // FFN: relu(x W1 + b1) W2 + b2 + x, then LN2 -> output
  k_gemm_wmma<1><<<dim3(16, 128), 256, 0, stream>>>(
      xh, W1_t, b1, nullptr, nullptr, Hh, nullptr, nullptr, nullptr, 16384, 2048, 512);
  k_gemm_wmma<2><<<dim3(4, 128), 256, 0, stream>>>(
      Hh, W2_t, b2, x32, pre, nullptr, nullptr, nullptr, nullptr, 16384, 512, 2048);
  k_layernorm<<<16384, 256, 0, stream>>>(pre, g2, be2, (float*)d_out, nullptr);
}
